// _Raindrop_2534030705296
// MI455X (gfx1250) — compile-verified, matrix operand order checked by reading, not run on hardware
//
#include <hip/hip_runtime.h>
#include <hip/hip_bf16.h>

// ---------------- model dims ----------------
#define T_     256
#define B_     32
#define N_     64
#define D_OB   4
#define D_T    272
#define D_TP   288      // D_T padded to multiple of 32 for WMMA K
#define H_     8
#define DH_    34
#define DHP    64       // head dim padded to multiple of 32
#define L_     2
#define DFF_   1024
#define FIN    1024     // T * D_OB

typedef __attribute__((ext_vector_type(16))) _Float16 v16h;
typedef __attribute__((ext_vector_type(8)))  _Float16 v8h;
typedef __attribute__((ext_vector_type(8)))  float    v8f;

// ---------------------------------------------------------------------------
// 16x32 f16 fragment load in the CDNA5 WMMA "A" layout from a row-major
// matrix with leading dimension ld (in elements).  Lane L (L<16) holds row
// row0+L, K = k0+g*8 .. +7 and k0+16+g*8 .. +7  (g = lane>>4).  Both chunks
// are 16-byte contiguous -> global_load_b128.  Row index clamped for edges.
// The same layout is used for B by loading from the pre-transposed weight
// (Bt is [N,K] row-major, lane = output column).
// ---------------------------------------------------------------------------
__device__ __forceinline__ v16h load_frag16x32(const _Float16* __restrict__ base,
                                               int ld, int row0, int k0, int rowMax) {
  int lane = threadIdx.x & 31;
  int r = row0 + (lane & 15);
  if (r > rowMax) r = rowMax;
  int g = lane >> 4;
  const _Float16* p = base + (size_t)r * ld + k0 + g * 8;
  v8h lo = *(const v8h*)p;
  v8h hi = *(const v8h*)(p + 16);
  v16h out;
#pragma unroll
  for (int i = 0; i < 8; ++i) { out[i] = lo[i]; out[i + 8] = hi[i]; }
  return out;
}

// ---------------------------------------------------------------------------
// Generic batched GEMM:  C = act(A @ B + bias)
//   A  : f16 [M, lda] row-major            (batch stride sA elements)
//   Bt : f16 [N, ldb] row-major == B^T     (batch stride sB elements)
//   C  : f32 [M, ldc] or f16 [M, ldc]      (batch stride sC elements)
// K must be a multiple of 32 (inputs zero-padded).  Block = 4 waves; each
// wave computes a 16x64 tile with 4 accumulators (4 v_wmma per K-step).
// ---------------------------------------------------------------------------
template <int RELU, int OUT16>
__global__ void gemm_bias_act(const _Float16* __restrict__ A,
                              const _Float16* __restrict__ Bt,
                              const float* __restrict__ bias,
                              float* __restrict__ C32,
                              _Float16* __restrict__ C16,
                              int M, int N, int K, int lda, int ldb, int ldc,
                              long long sA, long long sB, long long sC) {
  int batch = blockIdx.z;
  const _Float16* Ab = A + (size_t)batch * sA;
  const _Float16* Bb = Bt + (size_t)batch * sB;
  int wave = threadIdx.x >> 5;
  int lane = threadIdx.x & 31;
  int m0 = blockIdx.y * 64 + wave * 16;
  int n0 = blockIdx.x * 64;

  v8f acc0 = {}, acc1 = {}, acc2 = {}, acc3 = {};
  for (int k0 = 0; k0 < K; k0 += 32) {
    v16h a  = load_frag16x32(Ab, lda, m0,      k0, M - 1);
    v16h b0 = load_frag16x32(Bb, ldb, n0,      k0, N - 1);
    v16h b1 = load_frag16x32(Bb, ldb, n0 + 16, k0, N - 1);
    v16h b2 = load_frag16x32(Bb, ldb, n0 + 32, k0, N - 1);
    v16h b3 = load_frag16x32(Bb, ldb, n0 + 48, k0, N - 1);
    acc0 = __builtin_amdgcn_wmma_f32_16x16x32_f16(false, a, false, b0, (short)0, acc0, false, false);
    acc1 = __builtin_amdgcn_wmma_f32_16x16x32_f16(false, a, false, b1, (short)0, acc1, false, false);
    acc2 = __builtin_amdgcn_wmma_f32_16x16x32_f16(false, a, false, b2, (short)0, acc2, false, false);
    acc3 = __builtin_amdgcn_wmma_f32_16x16x32_f16(false, a, false, b3, (short)0, acc3, false, false);
  }

  v8f accs[4] = {acc0, acc1, acc2, acc3};
  int g = lane >> 4;
  int nl = lane & 15;
#pragma unroll
  for (int t = 0; t < 4; ++t) {
    int n = n0 + t * 16 + nl;
    if (n >= N) continue;
    float bv = bias ? bias[n] : 0.0f;
#pragma unroll
    for (int j = 0; j < 8; ++j) {
      int m = m0 + g * 8 + j;
      if (m >= M) continue;
      float v = accs[t][j] + bv;
      if (RELU) v = v > 0.f ? v : 0.f;
      if (OUT16) C16[(size_t)batch * sC + (size_t)m * ldc + n] = (_Float16)v;
      else       C32[(size_t)batch * sC + (size_t)m * ldc + n] = v;
    }
  }
}

// ---------------- elementwise / glue kernels ----------------

// A0[b*64+n, t*4+d] = f16(relu(x[t,b,n] * R_u[n*4+d]))   -> [2048, 1024]
__global__ void k_embed(const float* __restrict__ x, const float* __restrict__ R_u,
                        _Float16* __restrict__ A0) {
  int idx = blockIdx.x * blockDim.x + threadIdx.x;
  if (idx >= 2048 * FIN) return;
  int col = idx & (FIN - 1);
  int row = idx >> 10;
  int b = row >> 6, n = row & 63;
  int t = col >> 2, d = col & 3;
  float v = x[(t * B_ + b) * N_ + n] * R_u[n * 4 + d];
  A0[idx] = (_Float16)(v > 0.f ? v : 0.f);
}

// Wt[n, kpad] = f16(W[k, n]) zero-padded for k >= K
__global__ void k_wT(const float* __restrict__ W, _Float16* __restrict__ Wt,
                     int K, int N, int Kpad) {
  int idx = blockIdx.x * blockDim.x + threadIdx.x;
  if (idx >= N * Kpad) return;
  int n = idx / Kpad, k = idx - n * Kpad;
  Wt[idx] = (k < K) ? (_Float16)W[k * N + n] : (_Float16)0.f;
}

// m1[b,f] = mean over 64 nodes of C1[(b*64+n), f]  (uniform softmax gamma = 1/64)
__global__ void k_mean_nodes(const float* __restrict__ C1, _Float16* __restrict__ m1) {
  int idx = blockIdx.x * blockDim.x + threadIdx.x;
  if (idx >= B_ * FIN) return;
  int b = idx >> 10, f = idx & (FIN - 1);
  float s = 0.f;
  for (int n = 0; n < N_; ++n) s += C1[((b << 6) + n) * FIN + f];
  m1[idx] = (_Float16)(s * (1.0f / 64.0f));
}

// xt[t*B+b, :] :  cols 0..255 = v2[b, t*4 + (c&3)] ; 256..271 = sin/cos PE ; pad 0
__global__ void k_assemble_xt(const float* __restrict__ v2, const float* __restrict__ times,
                              float* __restrict__ xt32, _Float16* __restrict__ xt16) {
  int idx = blockIdx.x * blockDim.x + threadIdx.x;
  if (idx >= T_ * B_ * D_TP) return;
  int r = idx / D_TP, c = idx - r * D_TP;
  int t = r / B_, b = r - t * B_;
  if (c >= D_T) { xt16[idx] = (_Float16)0.f; return; }
  float val;
  if (c < 256) {
    val = v2[b * FIN + ((t << 2) | (c & 3))];
  } else {
    int i = c - 256;
    float inv = exp2f(-8.0f * (float)(i & 7) / 7.0f);  // 1 / 256^(i/7)
    float arg = times[t * B_ + b] * inv;
    val = (i < 8) ? sinf(arg) : cosf(arg);
  }
  xt16[idx] = (_Float16)val;
  xt32[(size_t)r * D_T + c] = val;
}

// split qkv -> padded per-head Q (scaled), K, and V^T
__global__ void k_scatter_heads(const float* __restrict__ qkv,
                                _Float16* __restrict__ Qh, _Float16* __restrict__ Kh,
                                _Float16* __restrict__ Vt) {
  int idx = blockIdx.x * blockDim.x + threadIdx.x;   // [b][h][t][dp]
  if (idx >= B_ * H_ * T_ * DHP) return;
  int dp = idx & 63;
  int t  = (idx >> 6) & 255;
  int h  = (idx >> 14) & 7;
  int b  = idx >> 17;
  float q = 0.f, k = 0.f, v = 0.f;
  if (dp < DH_) {
    const float* row = qkv + (size_t)(t * B_ + b) * (3 * D_T);
    q = row[h * DH_ + dp] * 0.17149858514250882f;  // 1/sqrt(34)
    k = row[D_T + h * DH_ + dp];
    v = row[2 * D_T + h * DH_ + dp];
  }
  Qh[idx] = (_Float16)q;
  Kh[idx] = (_Float16)k;
  Vt[(((size_t)(b * H_ + h)) * DHP + dp) * T_ + t] = (_Float16)v;
}

// masked row softmax: one wave per (b,h,q) row of 256 scores
__global__ void k_softmax(const float* __restrict__ scores, const int* __restrict__ lengths,
                          _Float16* __restrict__ attn) {
  int row = blockIdx.x * 4 + (threadIdx.x >> 5);
  int lane = threadIdx.x & 31;
  if (row >= B_ * H_ * T_) return;
  int b = row >> 11;
  int len = lengths[b];
  const float* s = scores + (size_t)row * T_;
  _Float16* a = attn + (size_t)row * T_;
  float vals[8];
  float mx = -1e30f;
#pragma unroll
  for (int i = 0; i < 8; ++i) {
    int c = lane + i * 32;
    float v = (c < len) ? s[c] : -1e30f;
    vals[i] = v;
    mx = fmaxf(mx, v);
  }
  for (int off = 16; off; off >>= 1) mx = fmaxf(mx, __shfl_xor(mx, off, 32));
  float sum = 0.f;
#pragma unroll
  for (int i = 0; i < 8; ++i) {
    float e = (vals[i] > -1e29f) ? __expf(vals[i] - mx) : 0.f;
    vals[i] = e;
    sum += e;
  }
  for (int off = 16; off; off >>= 1) sum += __shfl_xor(sum, off, 32);
  float inv = 1.0f / sum;
#pragma unroll
  for (int i = 0; i < 8; ++i) a[lane + i * 32] = (_Float16)(vals[i] * inv);
}

// heads back to [t*B+b, 288] f16 (pad 0)
__global__ void k_gather_heads(const float* __restrict__ AV, _Float16* __restrict__ o16) {
  int idx = blockIdx.x * blockDim.x + threadIdx.x;
  if (idx >= T_ * B_ * D_TP) return;
  int r = idx / D_TP, c = idx - r * D_TP;
  float v = 0.f;
  if (c < D_T) {
    int h = c / DH_, d = c - h * DH_;
    int t = r / B_, b = r - t * B_;
    v = AV[(((size_t)(b * H_ + h)) * T_ + t) * DHP + d];
  }
  o16[idx] = (_Float16)v;
}

// xt = LayerNorm(xt + res) * w + b  (wave per row), updating f32 + padded f16
__global__ void k_add_ln(const float* __restrict__ res,
                         const float* __restrict__ w, const float* __restrict__ bias,
                         float* __restrict__ xt32, _Float16* __restrict__ xt16) {
  int row = blockIdx.x * (blockDim.x >> 5) + (threadIdx.x >> 5);
  int lane = threadIdx.x & 31;
  if (row >= T_ * B_) return;
  float* xr = xt32 + (size_t)row * D_T;
  const float* rr = res + (size_t)row * D_T;
  float s = 0.f, s2 = 0.f;
  for (int c = lane; c < D_T; c += 32) {
    float v = xr[c] + rr[c];
    s += v; s2 += v * v;
  }
  for (int off = 16; off; off >>= 1) {
    s  += __shfl_xor(s, off, 32);
    s2 += __shfl_xor(s2, off, 32);
  }
  float mu   = s * (1.0f / D_T);
  float var  = s2 * (1.0f / D_T) - mu * mu;
  float rstd = rsqrtf(var + 1e-5f);
  for (int c = lane; c < D_T; c += 32) {
    float v = (xr[c] + rr[c] - mu) * rstd * w[c] + bias[c];
    xr[c] = v;
    xt16[(size_t)row * D_TP + c] = (_Float16)v;
  }
}

// masked mean over time:  agg[b,c] = sum_{t<len} xt32[t,b,c] / (len+1)
__global__ void k_agg(const float* __restrict__ xt32, const int* __restrict__ lengths,
                      float* __restrict__ agg) {
  int idx = blockIdx.x * blockDim.x + threadIdx.x;
  if (idx >= B_ * D_T) return;
  int b = idx / D_T, c = idx - b * D_T;
  int len = lengths[b];
  float s = 0.f;
  for (int t = 0; t < len; ++t) s += xt32[((size_t)t * B_ + b) * D_T + c];
  agg[idx] = s / ((float)len + 1.0f);
}

__global__ void k_mlp1(const float* __restrict__ agg, const float* __restrict__ w1,
                       const float* __restrict__ b1, float* __restrict__ hid) {
  int idx = blockIdx.x * blockDim.x + threadIdx.x;
  if (idx >= B_ * D_T) return;
  int b = idx / D_T, j = idx - b * D_T;
  float s = b1[j];
  for (int i = 0; i < D_T; ++i) s += agg[b * D_T + i] * w1[i * D_T + j];
  hid[idx] = s > 0.f ? s : 0.f;
}

__global__ void k_mlp2(const float* __restrict__ hid, const float* __restrict__ w2,
                       const float* __restrict__ b2, float* __restrict__ out) {
  int b = blockIdx.x * blockDim.x + threadIdx.x;
  if (b >= B_) return;
  float l0 = b2[0], l1 = b2[1];
  for (int i = 0; i < D_T; ++i) {
    float h = hid[b * D_T + i];
    l0 += h * w2[i * 2 + 0];
    l1 += h * w2[i * 2 + 1];
  }
  float m = fmaxf(l0, l1);
  float e0 = __expf(l0 - m), e1 = __expf(l1 - m);
  float inv = 1.0f / (e0 + e1);
  out[b * 2 + 0] = e0 * inv;
  out[b * 2 + 1] = e1 * inv;
}

// ---------------------------------------------------------------------------
extern "C" void kernel_launch(void* const* d_in, const int* in_sizes, int n_in,
                              void* d_out, int out_size, void* d_ws, size_t ws_size,
                              hipStream_t stream) {
  (void)in_sizes; (void)n_in; (void)out_size; (void)ws_size;

  const float* x      = (const float*)d_in[0];
  const float* times  = (const float*)d_in[1];
  const int*   lens   = (const int*)d_in[2];
  const float* R_u    = (const float*)d_in[3];
  const float* Wv1    = (const float*)d_in[4];
  const float* bv1    = (const float*)d_in[5];
  const float* Wv2    = (const float*)d_in[6];
  const float* bv2    = (const float*)d_in[7];
  const float* qkv_w  = (const float*)d_in[8];
  const float* qkv_b  = (const float*)d_in[9];
  const float* out_w  = (const float*)d_in[10];
  const float* out_b  = (const float*)d_in[11];
  const float* ln1_w  = (const float*)d_in[12];
  const float* ln1_b  = (const float*)d_in[13];
  const float* ln2_w  = (const float*)d_in[14];
  const float* ln2_b  = (const float*)d_in[15];
  const float* ffn_w1 = (const float*)d_in[16];
  const float* ffn_b1 = (const float*)d_in[17];
  const float* ffn_w2 = (const float*)d_in[18];
  const float* ffn_b2 = (const float*)d_in[19];
  const float* mlp_w1 = (const float*)d_in[20];
  const float* mlp_b1 = (const float*)d_in[21];
  const float* mlp_w2 = (const float*)d_in[22];
  const float* mlp_b2 = (const float*)d_in[23];

  // ---- workspace carve-up ----
  char* wsb = (char*)d_ws;
  size_t off = 0;
  auto alloc = [&](size_t bytes) -> void* {
    void* p = wsb + off;
    off = (off + bytes + 255) & ~(size_t)255;
    return p;
  };
  _Float16* A0    = (_Float16*)alloc((size_t)2048 * FIN * 2);
  _Float16* W1t   = (_Float16*)alloc((size_t)FIN * FIN * 2);
  _Float16* W2t   = (_Float16*)alloc((size_t)FIN * FIN * 2);
  float*    C1    = (float*)   alloc((size_t)2048 * FIN * 4);
  _Float16* m1    = (_Float16*)alloc((size_t)B_ * FIN * 2);
  float*    v2f   = (float*)   alloc((size_t)B_ * FIN * 4);
  float*    xt32  = (float*)   alloc((size_t)T_ * B_ * D_T * 4);
  _Float16* xt16  = (_Float16*)alloc((size_t)T_ * B_ * D_TP * 2);
  _Float16* Wqkvt = (_Float16*)alloc((size_t)816 * D_TP * 2);
  _Float16* Woutt = (_Float16*)alloc((size_t)D_T * D_TP * 2);
  _Float16* Wff1t = (_Float16*)alloc((size_t)DFF_ * D_TP * 2);
  _Float16* Wff2t = (_Float16*)alloc((size_t)D_T * DFF_ * 2);
  float*    qkvf  = (float*)   alloc((size_t)T_ * B_ * 3 * D_T * 4);
  _Float16* Qh    = (_Float16*)alloc((size_t)B_ * H_ * T_ * DHP * 2);
  _Float16* Kh    = (_Float16*)alloc((size_t)B_ * H_ * T_ * DHP * 2);
  _Float16* Vt    = (_Float16*)alloc((size_t)B_ * H_ * DHP * T_ * 2);
  float*    scrs  = (float*)   alloc((size_t)B_ * H_ * T_ * T_ * 4);
  _Float16* attn  = (_Float16*)alloc((size_t)B_ * H_ * T_ * T_ * 2);
  float*    AV    = (float*)   alloc((size_t)B_ * H_ * T_ * DHP * 4);
  _Float16* o16   = (_Float16*)alloc((size_t)T_ * B_ * D_TP * 2);
  float*    proj  = (float*)   alloc((size_t)T_ * B_ * D_T * 4);
  _Float16* ff16  = (_Float16*)alloc((size_t)T_ * B_ * DFF_ * 2);
  float*    aggf  = (float*)   alloc((size_t)B_ * D_T * 4);
  float*    hidf  = (float*)   alloc((size_t)B_ * D_T * 4);

  auto gemm = [&](const _Float16* A, const _Float16* Bt, const float* bias,
                  float* C32, _Float16* C16, int M, int N, int K,
                  int lda, int ldb, int ldc,
                  long long sA, long long sB, long long sC, int batch,
                  bool relu, bool out16) {
    dim3 grid((N + 63) / 64, (M + 63) / 64, batch);
    dim3 block(128);
    if (relu) {
      if (out16) gemm_bias_act<1, 1><<<grid, block, 0, stream>>>(A, Bt, bias, C32, C16, M, N, K, lda, ldb, ldc, sA, sB, sC);
      else       gemm_bias_act<1, 0><<<grid, block, 0, stream>>>(A, Bt, bias, C32, C16, M, N, K, lda, ldb, ldc, sA, sB, sC);
    } else {
      if (out16) gemm_bias_act<0, 1><<<grid, block, 0, stream>>>(A, Bt, bias, C32, C16, M, N, K, lda, ldb, ldc, sA, sB, sC);
      else       gemm_bias_act<0, 0><<<grid, block, 0, stream>>>(A, Bt, bias, C32, C16, M, N, K, lda, ldb, ldc, sA, sB, sC);
    }
  };

  // ---- observation embedding + ObservationPropagation (collapsed) ----
  k_embed<<<(2048 * FIN + 255) / 256, 256, 0, stream>>>(x, R_u, A0);
  k_wT<<<(FIN * FIN + 255) / 256, 256, 0, stream>>>(Wv1, W1t, FIN, FIN, FIN);
  k_wT<<<(FIN * FIN + 255) / 256, 256, 0, stream>>>(Wv2, W2t, FIN, FIN, FIN);
  gemm(A0, W1t, bv1, C1, nullptr, 2048, FIN, FIN, FIN, FIN, FIN, 0, 0, 0, 1, true, false);
  k_mean_nodes<<<(B_ * FIN + 255) / 256, 256, 0, stream>>>(C1, m1);
  gemm(m1, W2t, bv2, v2f, nullptr, B_, FIN, FIN, FIN, FIN, FIN, 0, 0, 0, 1, true, false);
  k_assemble_xt<<<(T_ * B_ * D_TP + 255) / 256, 256, 0, stream>>>(v2f, times, xt32, xt16);

  // ---- 2 transformer encoder layers ----
  const int TB = T_ * B_;           // 8192 rows
  const int BH = B_ * H_;           // 256 attention batches
  for (int l = 0; l < L_; ++l) {
    k_wT<<<(816 * D_TP + 255) / 256, 256, 0, stream>>>(qkv_w + (size_t)l * D_T * 816, Wqkvt, D_T, 816, D_TP);
    k_wT<<<(D_T * D_TP + 255) / 256, 256, 0, stream>>>(out_w + (size_t)l * D_T * D_T, Woutt, D_T, D_T, D_TP);
    k_wT<<<(DFF_ * D_TP + 255) / 256, 256, 0, stream>>>(ffn_w1 + (size_t)l * D_T * DFF_, Wff1t, D_T, DFF_, D_TP);
    k_wT<<<(D_T * DFF_ + 255) / 256, 256, 0, stream>>>(ffn_w2 + (size_t)l * DFF_ * D_T, Wff2t, DFF_, D_T, DFF_);

    // QKV projection: [8192,288] x [816,288]^T
    gemm(xt16, Wqkvt, qkv_b + l * 816, qkvf, nullptr, TB, 816, D_TP, D_TP, D_TP, 816, 0, 0, 0, 1, false, false);
    k_scatter_heads<<<(B_ * H_ * T_ * DHP + 255) / 256, 256, 0, stream>>>(qkvf, Qh, Kh, Vt);

    // scores = (Q/sqrt(dh)) @ K^T   batched over B*H
    gemm(Qh, Kh, nullptr, scrs, nullptr, T_, T_, DHP, DHP, DHP, T_,
         (long long)T_ * DHP, (long long)T_ * DHP, (long long)T_ * T_, BH, false, false);
    k_softmax<<<(B_ * H_ * T_) / 4, 128, 0, stream>>>(scrs, lens, attn);

    // attn @ V   batched
    gemm(attn, Vt, nullptr, AV, nullptr, T_, DHP, T_, T_, T_, DHP,
         (long long)T_ * T_, (long long)DHP * T_, (long long)T_ * DHP, BH, false, false);
    k_gather_heads<<<(TB * D_TP + 255) / 256, 256, 0, stream>>>(AV, o16);

    // output projection + residual + LN1
    gemm(o16, Woutt, out_b + l * D_T, proj, nullptr, TB, D_T, D_TP, D_TP, D_TP, D_T, 0, 0, 0, 1, false, false);
    k_add_ln<<<(TB + 3) / 4, 128, 0, stream>>>(proj, ln1_w + l * D_T, ln1_b + l * D_T, xt32, xt16);

    // FFN
    gemm(xt16, Wff1t, ffn_b1 + l * DFF_, nullptr, ff16, TB, DFF_, D_TP, D_TP, D_TP, DFF_, 0, 0, 0, 1, true, true);
    gemm(ff16, Wff2t, ffn_b2 + l * D_T, proj, nullptr, TB, D_T, DFF_, DFF_, DFF_, D_T, 0, 0, 0, 1, false, false);
    k_add_ln<<<(TB + 3) / 4, 128, 0, stream>>>(proj, ln2_w + l * D_T, ln2_b + l * D_T, xt32, xt16);
  }

  // ---- masked mean pool + classifier head ----
  k_agg<<<(B_ * D_T + 255) / 256, 256, 0, stream>>>(xt32, lens, aggf);
  k_mlp1<<<(B_ * D_T + 255) / 256, 256, 0, stream>>>(aggf, mlp_w1, mlp_b1, hidf);
  k_mlp2<<<1, 32, 0, stream>>>(hidf, mlp_w2, mlp_b2, (float*)d_out);
}
